// HGNNConv2_18348100288558
// MI455X (gfx1250) — matrix operand order repaired and checked by reading, not compiled, not run on hardware
//
#include <hip/hip_runtime.h>

// Problem constants (match reference)
#define N_V 16384   // vertices
#define M_E 8192    // hyperedges
#define CIN 128
#define COUT 64

typedef __attribute__((ext_vector_type(16))) __bf16 v16bf;
typedef __attribute__((ext_vector_type(8)))  float  v8f;

union FragB16 {
    v16bf v;
    uint4 q[2];
    unsigned int u[8];
    unsigned short s[16];
};

// Round-to-nearest-even f32 -> bf16 (for real-valued data)
static __device__ __forceinline__ unsigned short f2bf_rne(float f) {
    unsigned int u = __float_as_uint(f);
    u += 0x7FFFu + ((u >> 16) & 1u);
    return (unsigned short)(u >> 16);
}
// Truncating pack (exact for 0.0f / 1.0f incidence values)
static __device__ __forceinline__ unsigned int pack2bf_trunc(float lo, float hi) {
    return (__float_as_uint(lo) >> 16) | (__float_as_uint(hi) & 0xFFFF0000u);
}

// ---------------------------------------------------------------------------
// Kernel 1: dv[n] = rowsum(H[n,:])^{-1/2}  (nan_to_num -> 0 for empty rows)
// ---------------------------------------------------------------------------
__global__ void __launch_bounds__(256)
k_rowsum_dv(const float* __restrict__ H, float* __restrict__ dv) {
    __shared__ float red[256];
    const int row = blockIdx.x;
    const float4* hp = reinterpret_cast<const float4*>(H + (size_t)row * M_E);
    float s = 0.f;
    for (int i = threadIdx.x; i < M_E / 4; i += 256) {
        float4 v = hp[i];
        s += v.x + v.y + v.z + v.w;
    }
    red[threadIdx.x] = s;
    __syncthreads();
    for (int off = 128; off > 0; off >>= 1) {
        if (threadIdx.x < off) red[threadIdx.x] += red[threadIdx.x + off];
        __syncthreads();
    }
    if (threadIdx.x == 0) {
        float t = red[0];
        dv[row] = (t > 0.f) ? rsqrtf(t) : 0.f;
    }
}

// ---------------------------------------------------------------------------
// Kernel 2: PT[c][n] = bf16( dv[n] * (X @ W^T + b)[n][c] ), stored transposed
// ---------------------------------------------------------------------------
__global__ void __launch_bounds__(64)
k_project(const float* __restrict__ X, const float* __restrict__ W,
          const float* __restrict__ b, const float* __restrict__ dv,
          unsigned short* __restrict__ PT) {
    __shared__ float xrow[CIN];
    const int n = blockIdx.x;
    const int c = threadIdx.x;                 // 64 threads = COUT
    xrow[c]      = X[(size_t)n * CIN + c];
    xrow[c + 64] = X[(size_t)n * CIN + c + 64];
    __syncthreads();
    const float* w = W + (size_t)c * CIN;
    float acc = b[c];
#pragma unroll 8
    for (int k = 0; k < CIN; ++k) acc += xrow[k] * w[k];
    PT[(size_t)c * N_V + n] = f2bf_rne(acc * dv[n]);
}

// ---------------------------------------------------------------------------
// Kernel 3: GEMM1  Y1T[c][m] = bf16( de[m] * sum_n H[n][m] * P[n][c] )
//   - one block per 16-wide m-tile, 4 waves = 4 c-tiles of 16
//   - 64 H rows staged per iter into LDS, transposed + bf16-converted
//   - column sums of H fused (computed over the same loads) -> de in-block
// ---------------------------------------------------------------------------
#define AT_STRIDE 72   // 16 rows x (64 + 8 pad) bf16 elements
__global__ void __launch_bounds__(128)
k_gemm1(const float* __restrict__ H, const unsigned short* __restrict__ PT,
        unsigned short* __restrict__ Y1T) {
    __shared__ unsigned short At[16 * AT_STRIDE]; // At[m'][n'], n' in [0,64)
    __shared__ float csum[16];

    const int tid  = threadIdx.x;
    const int m0   = blockIdx.x * 16;
    const int quad = tid & 3;     // column group: cols m0 + quad*4 .. +3
    const int rowi = tid >> 2;    // staging rows rowi and rowi+32
    const int wave = tid >> 5;    // c-tile
    const int lane = tid & 31;
    const int mrow = lane & 15;
    const int half = lane >> 4;

    if (tid < 16) csum[tid] = 0.f;
    float4 cacc = make_float4(0.f, 0.f, 0.f, 0.f);
    v8f acc = {};
    __syncthreads();

    for (int n0 = 0; n0 < N_V; n0 += 64) {
        // ---- stage 64x16 f32 tile of H, write transposed bf16 into LDS ----
#pragma unroll
        for (int rr = 0; rr < 2; ++rr) {
            const int r = rowi + rr * 32;
            float4 hv = *reinterpret_cast<const float4*>(
                H + (size_t)(n0 + r) * M_E + m0 + quad * 4);
            cacc.x += hv.x; cacc.y += hv.y; cacc.z += hv.z; cacc.w += hv.w;
            At[(quad * 4 + 0) * AT_STRIDE + r] = (unsigned short)(__float_as_uint(hv.x) >> 16);
            At[(quad * 4 + 1) * AT_STRIDE + r] = (unsigned short)(__float_as_uint(hv.y) >> 16);
            At[(quad * 4 + 2) * AT_STRIDE + r] = (unsigned short)(__float_as_uint(hv.z) >> 16);
            At[(quad * 4 + 3) * AT_STRIDE + r] = (unsigned short)(__float_as_uint(hv.w) >> 16);
        }
        __syncthreads();

        // ---- two K=32 WMMA steps over the staged tile ----
#pragma unroll
        for (int s = 0; s < 2; ++s) {
            FragB16 a, bb;
            const int nbase = s * 32 + half * 8;
            a.q[0] = *reinterpret_cast<const uint4*>(&At[mrow * AT_STRIDE + nbase]);
            a.q[1] = *reinterpret_cast<const uint4*>(&At[mrow * AT_STRIDE + nbase + 16]);
            const unsigned short* bp =
                PT + (size_t)(wave * 16 + mrow) * N_V + n0 + s * 32 + half * 8;
            bb.q[0] = *reinterpret_cast<const uint4*>(bp);
            bb.q[1] = *reinterpret_cast<const uint4*>(bp + 16);
            acc = __builtin_amdgcn_wmma_f32_16x16x32_bf16(
                false, a.v, false, bb.v, (short)0, acc, false, false);
        }
        __syncthreads();
    }

    // ---- fold per-thread column partials into LDS, finalize de in-block ----
    atomicAdd(&csum[quad * 4 + 0], cacc.x);
    atomicAdd(&csum[quad * 4 + 1], cacc.y);
    atomicAdd(&csum[quad * 4 + 2], cacc.z);
    atomicAdd(&csum[quad * 4 + 3], cacc.w);
    __syncthreads();

#pragma unroll
    for (int v = 0; v < 8; ++v) {
        const int mloc = v + 8 * half;          // C-layout: M = v + 8*(lane/16)
        const float s = csum[mloc];
        const float de = (s > 0.f) ? (1.0f / s) : 0.f;
        const int c = wave * 16 + mrow;         // N = lane % 16
        Y1T[(size_t)c * M_E + (m0 + mloc)] = f2bf_rne(acc[v] * de);
    }
}

// ---------------------------------------------------------------------------
// Kernel 4: GEMM2  out[n][c] = relu( dv[n] * sum_m H[n][m] * Y1e[m][c] )
//   - one block per 16-row n-tile, 4 waves = 4 c-tiles
//   - A fragments loaded directly from row-major f32 H, packed to bf16
// ---------------------------------------------------------------------------
__global__ void __launch_bounds__(128)
k_gemm2(const float* __restrict__ H, const unsigned short* __restrict__ Y1T,
        const float* __restrict__ dv, float* __restrict__ out) {
    const int tid  = threadIdx.x;
    const int wave = tid >> 5;
    const int lane = tid & 31;
    const int mrow = lane & 15;
    const int half = lane >> 4;
    const int n0   = blockIdx.x * 16;

    const float* hrow =
        H + (size_t)(n0 + mrow) * M_E;                 // this lane's H row
    const unsigned short* brow =
        Y1T + (size_t)(wave * 16 + mrow) * M_E;        // this lane's Y1T row
    v8f acc = {};

    for (int k = 0; k < M_E; k += 32) {
        __builtin_prefetch(hrow + k + 512, 0, 1);      // global_prefetch_b8
        FragB16 a, bb;
        // A: lanes<16 take K(+0..7,+16..23); lanes>=16 take K(+8..15,+24..31)
        const float4* ap0 = reinterpret_cast<const float4*>(hrow + k + half * 8);
        const float4* ap1 = reinterpret_cast<const float4*>(hrow + k + 16 + half * 8);
        float4 a0 = ap0[0], a1 = ap0[1];
        float4 a2 = ap1[0], a3 = ap1[1];
        a.u[0] = pack2bf_trunc(a0.x, a0.y);
        a.u[1] = pack2bf_trunc(a0.z, a0.w);
        a.u[2] = pack2bf_trunc(a1.x, a1.y);
        a.u[3] = pack2bf_trunc(a1.z, a1.w);
        a.u[4] = pack2bf_trunc(a2.x, a2.y);
        a.u[5] = pack2bf_trunc(a2.z, a2.w);
        a.u[6] = pack2bf_trunc(a3.x, a3.y);
        a.u[7] = pack2bf_trunc(a3.z, a3.w);
        // B: lanes<16 K(+0..7,+16..23); lanes>=16 K(+8..15,+24..31)
        const unsigned short* bp = brow + k + half * 8;
        bb.q[0] = *reinterpret_cast<const uint4*>(bp);
        bb.q[1] = *reinterpret_cast<const uint4*>(bp + 16);
        acc = __builtin_amdgcn_wmma_f32_16x16x32_bf16(
            false, a.v, false, bb.v, (short)0, acc, false, false);
    }

#pragma unroll
    for (int v = 0; v < 8; ++v) {
        const int n = n0 + v + 8 * half;
        float val = acc[v] * dv[n];
        val = fmaxf(val, 0.f);
        out[(size_t)n * COUT + wave * 16 + mrow] = val;
    }
}

// ---------------------------------------------------------------------------
extern "C" void kernel_launch(void* const* d_in, const int* in_sizes, int n_in,
                              void* d_out, int out_size, void* d_ws, size_t ws_size,
                              hipStream_t stream) {
    const float* X = (const float*)d_in[0];   // N_V x CIN
    const float* H = (const float*)d_in[1];   // N_V x M_E
    const float* W = (const float*)d_in[2];   // COUT x CIN
    const float* b = (const float*)d_in[3];   // COUT
    float* out = (float*)d_out;               // N_V x COUT

    // Workspace carve-out (~3.1 MB): all buffers fully written before read.
    char* ws = (char*)d_ws;
    unsigned short* PT  = (unsigned short*)ws;                              // 64 x N_V bf16 (2 MB)
    unsigned short* Y1T = (unsigned short*)(ws + (size_t)COUT * N_V * 2);   // 64 x M_E bf16 (1 MB)
    float* dv = (float*)(ws + (size_t)COUT * N_V * 2 + (size_t)COUT * M_E * 2); // N_V f32

    k_rowsum_dv<<<N_V, 256, 0, stream>>>(H, dv);
    k_project  <<<N_V, 64, 0, stream>>>(X, W, b, dv, PT);
    k_gemm1    <<<M_E / 16, 128, 0, stream>>>(H, PT, Y1T);
    k_gemm2    <<<N_V / 16, 128, 0, stream>>>(H, Y1T, dv, out);
}
